// GAT_Framework_67070209294554
// MI455X (gfx1250) — matrix operand order repaired and checked by reading, not compiled, not run on hardware
//
#include <hip/hip_runtime.h>

typedef __attribute__((ext_vector_type(16))) _Float16 v16h;
typedef __attribute__((ext_vector_type(8)))  _Float16 v8h;
typedef __attribute__((ext_vector_type(4)))  _Float16 v4h;
typedef __attribute__((ext_vector_type(8)))  float    v8f;

#define N_NODES   50000
#define IN_FEATS  256
#define HEADS     4
#define NHID      64
#define NCLS      40
#define NEG_SLOPE 0.2f

#define GEMM_K    256          // K is 256 for all three layers
#define SB_STRIDE 40           // padded col stride (halves): 80B keeps 16B align, spreads banks

// ---------------------------------------------------------------------------
// row-normalize x (50000 x 256) and convert to f16
// ---------------------------------------------------------------------------
__global__ void __launch_bounds__(256)
normalize_h16(const float* __restrict__ x, _Float16* __restrict__ h16) {
    __shared__ float red[256];
    const int n = blockIdx.x;
    const int t = threadIdx.x;
    float v = x[(size_t)n * IN_FEATS + t];
    red[t] = v;
    __syncthreads();
    for (int s = 128; s > 0; s >>= 1) {
        if (t < s) red[t] += red[t + s];
        __syncthreads();
    }
    float denom = fmaxf(red[0], 1.0f);
    h16[(size_t)n * IN_FEATS + t] = (_Float16)(v / denom);
}

__global__ void convert_f16(const float* __restrict__ w, _Float16* __restrict__ w16, int n) {
    int i = blockIdx.x * blockDim.x + threadIdx.x;
    int stride = gridDim.x * blockDim.x;
    for (; i < n; i += stride) w16[i] = (_Float16)w[i];
}

__global__ void fill_f32(float* __restrict__ p, size_t n, float v) {
    size_t i = (size_t)blockIdx.x * blockDim.x + threadIdx.x;
    size_t stride = (size_t)gridDim.x * blockDim.x;
    for (; i < n; i += stride) p[i] = v;
}

// ---------------------------------------------------------------------------
// WMMA GEMM: C[M,N] = A[M,K]*B[K,N], M=50000(=3125*16), K=256, N=256|160
// block = 128 threads = 4 waves; block tile = 16 rows x 128 cols.
// A strip (16x256) staged once; B staged per 32-K chunk in COLUMN-MAJOR LDS
// so each lane's WMMA fragment is two contiguous ds_load_b128 reads.
// ---------------------------------------------------------------------------
template <int N>
__global__ void __launch_bounds__(128)
gemm_h16_wmma(const _Float16* __restrict__ A, const _Float16* __restrict__ B,
              float* __restrict__ C) {
    __shared__ _Float16 sA[16 * GEMM_K];        // 8 KB, row-major
    __shared__ _Float16 sB[128 * SB_STRIDE];    // 10 KB, column-major padded

    const int tid  = threadIdx.x;
    const int wave = tid >> 5;
    const int lane = tid & 31;
    const int rowbase  = blockIdx.x << 4;
    const int colblock = blockIdx.y << 7;
    const int row = lane & 15;                  // A row / C col within tile
    const int kh  = lane >> 4;                  // half-wave selector (ISA layout)
    const int cl0 = wave * 32 + (lane & 15);    // local col, tile 2*wave
    const int cl1 = cl0 + 16;                   // local col, tile 2*wave+1

    // ---- stage A strip 16x256 once: 4096 halves = 128 thr x 4 x v8h ----
    #pragma unroll
    for (int it = 0; it < 4; ++it) {
        int flat = (it << 10) + (tid << 3);     // step 1024 halves
        int r = flat >> 8, c = flat & 255;
        *(v8h*)&sA[flat] = *(const v8h*)&A[(size_t)(rowbase + r) * GEMM_K + c];
    }

    v8f acc0 = {}; v8f acc1 = {};
    union Frag { v16h v; v8h h[2]; };

    #pragma unroll
    for (int k0 = 0; k0 < GEMM_K; k0 += 32) {
        // ---- stage B chunk 32 x 128 into column-major LDS ----
        // thread loads v4h of row r, cols c..c+3; scatters as 4 u16 stores
        #pragma unroll
        for (int it = 0; it < 8; ++it) {
            int flat = (it << 9) + (tid << 2);
            int r = flat >> 7, c = flat & 127;
            v4h v = {};
            if (colblock + c < N)
                v = *(const v4h*)&B[(size_t)(k0 + r) * N + colblock + c];
            sB[(c + 0) * SB_STRIDE + r] = v[0];
            sB[(c + 1) * SB_STRIDE + r] = v[1];
            sB[(c + 2) * SB_STRIDE + r] = v[2];
            sB[(c + 3) * SB_STRIDE + r] = v[3];
        }
        if (k0 + 32 < GEMM_K) {   // prefetch next B chunk (global_prefetch_b8)
            __builtin_prefetch(&B[(size_t)(k0 + 32 + (tid >> 2)) * N + colblock], 0, 3);
        }
        __syncthreads();

        // ---- A fragment: two contiguous 16B LDS reads ----
        Frag a;
        a.h[0] = *(const v8h*)&sA[row * GEMM_K + k0 + (kh << 4)];
        a.h[1] = *(const v8h*)&sA[row * GEMM_K + k0 + (kh << 4) + 8];

        // ---- B fragments: column-major -> two contiguous 16B LDS reads each ----
        Frag b0, b1;
        b0.h[0] = *(const v8h*)&sB[cl0 * SB_STRIDE + (kh << 4)];
        b0.h[1] = *(const v8h*)&sB[cl0 * SB_STRIDE + (kh << 4) + 8];
        b1.h[0] = *(const v8h*)&sB[cl1 * SB_STRIDE + (kh << 4)];
        b1.h[1] = *(const v8h*)&sB[cl1 * SB_STRIDE + (kh << 4) + 8];

        acc0 = __builtin_amdgcn_wmma_f32_16x16x32_f16(false, a.v, false, b0.v,
                                                      (short)0, acc0, false, false);
        acc1 = __builtin_amdgcn_wmma_f32_16x16x32_f16(false, a.v, false, b1.v,
                                                      (short)0, acc1, false, false);
        __syncthreads();
    }

    // C layout: VGPR r -> M = r + kh*8 ; N = lane%16
    #pragma unroll
    for (int r = 0; r < 8; ++r) {
        int m  = r + (kh << 3);
        int n0 = colblock + cl0;
        int n1 = colblock + cl1;
        if (n0 < N) C[(size_t)(rowbase + m) * N + n0] = acc0[r];
        if (n1 < N) C[(size_t)(rowbase + m) * N + n1] = acc1[r];
    }
}

// ---------------------------------------------------------------------------
// el/er per (node, head):  el[n,h] = sum_f feat[n,h,f]*al[h,f]
// ---------------------------------------------------------------------------
__global__ void attn_scores(const float* __restrict__ feat,
                            const float* __restrict__ al, const float* __restrict__ ar,
                            float* __restrict__ el, float* __restrict__ er, int F) {
    int idx = blockIdx.x * blockDim.x + threadIdx.x;
    if (idx >= N_NODES * HEADS) return;
    int n = idx / HEADS, h = idx % HEADS;
    const float* f = feat + (size_t)n * HEADS * F + h * F;
    float sl = 0.f, sr = 0.f;
    for (int i = 0; i < F; ++i) {
        float v = f[i];
        sl += v * al[h * F + i];
        sr += v * ar[h * F + i];
    }
    el[idx] = sl;
    er[idx] = sr;
}

__device__ inline void atomicMaxF(float* addr, float val) {
    if (val >= 0.0f) atomicMax((int*)addr, __float_as_int(val));
    else             atomicMin((unsigned int*)addr, __float_as_uint(val));
}

__global__ void edge_max(const int* __restrict__ src, const int* __restrict__ dst,
                         const float* __restrict__ el, const float* __restrict__ er,
                         float* __restrict__ emax, int E) {
    int idx = blockIdx.x * blockDim.x + threadIdx.x;
    if (idx >= E * HEADS) return;
    int e = idx >> 2, h = idx & 3;
    float v = el[src[e] * HEADS + h] + er[dst[e] * HEADS + h];
    v = v > 0.f ? v : NEG_SLOPE * v;
    atomicMaxF(&emax[dst[e] * HEADS + h], v);
}

__global__ void edge_exp(const int* __restrict__ src, const int* __restrict__ dst,
                         const float* __restrict__ el, const float* __restrict__ er,
                         const float* __restrict__ emax, float* __restrict__ ee,
                         float* __restrict__ denom, int E) {
    int idx = blockIdx.x * blockDim.x + threadIdx.x;
    if (idx >= E * HEADS) return;
    int e = idx >> 2, h = idx & 3;
    int d = dst[e];
    float v = el[src[e] * HEADS + h] + er[d * HEADS + h];
    v = v > 0.f ? v : NEG_SLOPE * v;
    float x = __expf(v - emax[d * HEADS + h]);
    ee[idx] = x;
    atomicAdd(&denom[d * HEADS + h], x);
}

// scatter-sum:  agg[dst,h,f] += feat[src,h,f] * alpha[e,h]
__global__ void aggregate_k(const int* __restrict__ src, const int* __restrict__ dst,
                            const float* __restrict__ feat, const float* __restrict__ ee,
                            const float* __restrict__ denom, float* __restrict__ agg,
                            int E, int F) {
    long idx = (long)blockIdx.x * blockDim.x + threadIdx.x;
    if (idx >= (long)E * F) return;
    int e = (int)(idx / F), f = (int)(idx % F);
    int s = src[e], d = dst[e];
    #pragma unroll
    for (int h = 0; h < HEADS; ++h) {
        float alpha = ee[e * HEADS + h] / denom[d * HEADS + h];
        atomicAdd(&agg[(size_t)d * HEADS * F + h * F + f],
                  feat[(size_t)s * HEADS * F + h * F + f] * alpha);
    }
}

// layers 0/1: h_out = f16( elu(agg + bias) )
__global__ void finalize01(const float* __restrict__ agg, const float* __restrict__ bias,
                           _Float16* __restrict__ hout, size_t n, int W) {
    size_t i = (size_t)blockIdx.x * blockDim.x + threadIdx.x;
    if (i >= n) return;
    float v = agg[i] + bias[i % W];
    v = v > 0.f ? v : expm1f(v);
    hout[i] = (_Float16)v;
}

// layer 2: logits = mean over heads of (agg + bias)
__global__ void finalize2(const float* __restrict__ agg, const float* __restrict__ bias,
                          float* __restrict__ logits) {
    int idx = blockIdx.x * blockDim.x + threadIdx.x;
    if (idx >= N_NODES * NCLS) return;
    int n = idx / NCLS, c = idx % NCLS;
    float s = 0.f;
    #pragma unroll
    for (int h = 0; h < HEADS; ++h)
        s += agg[(size_t)n * HEADS * NCLS + h * NCLS + c] + bias[h * NCLS + c];
    logits[idx] = s * 0.25f;
}

__global__ void log_softmax_k(const float* __restrict__ logits, float* __restrict__ out) {
    int n = blockIdx.x * blockDim.x + threadIdx.x;
    if (n >= N_NODES) return;
    const float* r = logits + (size_t)n * NCLS;
    float m = -1e30f;
    for (int c = 0; c < NCLS; ++c) m = fmaxf(m, r[c]);
    float s = 0.f;
    for (int c = 0; c < NCLS; ++c) s += __expf(r[c] - m);
    float ls = __logf(s) + m;
    for (int c = 0; c < NCLS; ++c) out[(size_t)n * NCLS + c] = r[c] - ls;
}

// ---------------------------------------------------------------------------
extern "C" void kernel_launch(void* const* d_in, const int* in_sizes, int n_in,
                              void* d_out, int out_size, void* d_ws, size_t ws_size,
                              hipStream_t stream) {
    const float* x   = (const float*)d_in[0];
    const int*   src = (const int*)  d_in[1];
    const int*   dst = (const int*)  d_in[2];
    const float* Wl[3]  = {(const float*)d_in[3],  (const float*)d_in[7],  (const float*)d_in[11]};
    const float* all[3] = {(const float*)d_in[4],  (const float*)d_in[8],  (const float*)d_in[12]};
    const float* arl[3] = {(const float*)d_in[5],  (const float*)d_in[9],  (const float*)d_in[13]};
    const float* bl[3]  = {(const float*)d_in[6],  (const float*)d_in[10], (const float*)d_in[14]};
    const int E = in_sizes[1];

    // ---- workspace carve ----
    char* p = (char*)d_ws;
    auto carve = [&](size_t bytes) { char* r = p; p += (bytes + 255) & ~(size_t)255; return (void*)r; };
    _Float16* h16    = (_Float16*)carve((size_t)N_NODES * 256 * sizeof(_Float16));
    float*    feat   = (float*)   carve((size_t)N_NODES * 256 * sizeof(float));
    float*    agg    = (float*)   carve((size_t)N_NODES * 256 * sizeof(float));
    _Float16* w16    = (_Float16*)carve((size_t)256 * 256 * sizeof(_Float16));
    float*    el     = (float*)   carve((size_t)N_NODES * HEADS * sizeof(float));
    float*    er     = (float*)   carve((size_t)N_NODES * HEADS * sizeof(float));
    float*    emax   = (float*)   carve((size_t)N_NODES * HEADS * sizeof(float));
    float*    denom  = (float*)   carve((size_t)N_NODES * HEADS * sizeof(float));
    float*    ee     = (float*)   carve((size_t)E * HEADS * sizeof(float));
    float*    logits = (float*)   carve((size_t)N_NODES * NCLS * sizeof(float));

    const int K = GEMM_K;
    const int TPB = 256;
    auto cdiv = [](long a, long b) { return (int)((a + b - 1) / b); };

    // 0) normalize + f16
    normalize_h16<<<N_NODES, 256, 0, stream>>>(x, h16);

    for (int L = 0; L < 3; ++L) {
        const int Nw = (L == 2) ? HEADS * NCLS : HEADS * NHID;  // 160 or 256
        const int F  = (L == 2) ? NCLS : NHID;                  // 40 or 64

        // 1) convert weights
        convert_f16<<<cdiv(K * Nw, TPB), TPB, 0, stream>>>(Wl[L], w16, K * Nw);

        // 2) feat = h @ W  (WMMA)
        dim3 ggrid(N_NODES / 16, cdiv(Nw, 128));
        if (L == 2) gemm_h16_wmma<160><<<ggrid, 128, 0, stream>>>(h16, w16, feat);
        else        gemm_h16_wmma<256><<<ggrid, 128, 0, stream>>>(h16, w16, feat);

        // 3) attention scores
        attn_scores<<<cdiv((long)N_NODES * HEADS, TPB), TPB, 0, stream>>>(
            feat, all[L], arl[L], el, er, F);

        // 4) init reductions
        fill_f32<<<1024, TPB, 0, stream>>>(emax, (size_t)N_NODES * HEADS, -1e30f);
        fill_f32<<<1024, TPB, 0, stream>>>(denom, (size_t)N_NODES * HEADS, 0.f);
        fill_f32<<<4096, TPB, 0, stream>>>(agg, (size_t)N_NODES * Nw, 0.f);

        // 5) segment max / 6) exp + denom
        edge_max<<<cdiv((long)E * HEADS, TPB), TPB, 0, stream>>>(src, dst, el, er, emax, E);
        edge_exp<<<cdiv((long)E * HEADS, TPB), TPB, 0, stream>>>(src, dst, el, er, emax, ee, denom, E);

        // 7) scatter-sum messages
        aggregate_k<<<cdiv((long)E * F, TPB), TPB, 0, stream>>>(src, dst, feat, ee, denom, agg, E, F);

        // 8) finalize
        if (L < 2) {
            finalize01<<<cdiv((long)N_NODES * Nw, TPB), TPB, 0, stream>>>(
                agg, bl[L], h16, (size_t)N_NODES * Nw, Nw);
        } else {
            finalize2<<<cdiv((long)N_NODES * NCLS, TPB), TPB, 0, stream>>>(agg, bl[L], logits);
        }
    }

    // 9) log-softmax into d_out
    log_softmax_k<<<cdiv(N_NODES, TPB), TPB, 0, stream>>>(logits, (float*)d_out);
}
